// CARAFEModule_23115513987462
// MI455X (gfx1250) — matrix operand order repaired and checked by reading, not compiled, not run on hardware
//
#include <hip/hip_runtime.h>
#include <hip/hip_fp16.h>

typedef __attribute__((ext_vector_type(16))) _Float16 v16h;
typedef __attribute__((ext_vector_type(8)))  _Float16 v8h;
typedef __attribute__((ext_vector_type(8)))  float    v8f;

#define B_    4
#define C_    256
#define H_    64
#define W_    64
#define HW_   4096
#define CC_   64      // compressed channels
#define ENC_  196     // encoder output channels
#define ENCP_ 208     // padded to 13 * 16
#define K2_   49      // 7*7 carafe taps
#define KTAPS_ 25     // 5*5 encoder taps
#define KENC_ 1600    // 25 * 64 reduction length
#define OH_   128
#define OW_   128

// Encoder spatial tiling: 4 rows x 16 cols per block, 5x5 halo -> 8 x 20
#define TSY_  4
#define TSX_  16
#define THY_  8
#define THX_  20

// ---------------------------------------------------------------------------
// Weight prep: f32 -> f16, rearranged into WMMA B-matrix friendly layouts.
// B layout (16-bit, 32x16): lane l holds column N=l%16 and 16 K values, in the A K order
// => stored [K][N], so a lane's 16 halves are strided by N.
// ---------------------------------------------------------------------------
__global__ void k_prep_wcomp(const float* __restrict__ w_comp,
                             _Float16* __restrict__ wcomp16) {
    int idx = blockIdx.x * blockDim.x + threadIdx.x;
    if (idx >= C_ * CC_) return;
    int c = idx >> 6, oc = idx & 63;            // wcomp16[c][oc]
    wcomp16[idx] = (_Float16)w_comp[oc * C_ + c];
}

__global__ void k_prep_wenc(const float* __restrict__ w_enc,
                            _Float16* __restrict__ wenc16) {
    int idx = blockIdx.x * blockDim.x + threadIdx.x;
    if (idx >= KENC_ * ENCP_) return;
    int k = idx / ENCP_, n = idx % ENCP_;       // k = tap*64 + c
    int tap = k >> 6, c = k & 63;
    float v = (n < ENC_) ? w_enc[(size_t)n * KENC_ + c * KTAPS_ + tap] : 0.f;
    wenc16[idx] = (_Float16)v;
}

// ---------------------------------------------------------------------------
// LDS tile transpose: x [B,C,HW] f32 -> x16 [B,HW,C] f16 (channel-last)
// so WMMA A fragments become contiguous 16B runs per lane.
// ---------------------------------------------------------------------------
__global__ void k_transpose_x(const float* __restrict__ x,
                              _Float16* __restrict__ x16) {
    __shared__ _Float16 t[64][65];
    int blk = blockIdx.x;            // b(4) x hwTile(64) x cTile(4) = 1024
    int b   = blk >> 8;
    int rem = blk & 255;
    int ht  = rem >> 2;
    int ct  = rem & 3;
    int tid = threadIdx.x;
    for (int i = tid; i < 4096; i += 256) {
        int c = i >> 6, hwl = i & 63;           // coalesced read along hw
        t[c][hwl] = (_Float16)x[((size_t)(b * C_ + ct * 64 + c)) * HW_ + ht * 64 + hwl];
    }
    __syncthreads();
    for (int i = tid; i < 4096; i += 256) {
        int hwl = i >> 6, c = i & 63;           // coalesced write along c
        x16[((size_t)(b * HW_ + ht * 64 + hwl)) * C_ + ct * 64 + c] = t[c][hwl];
    }
}

// ---------------------------------------------------------------------------
// Kernel 1: 1x1 compress conv as GEMM  M=16384 (pixels) N=64 K=256.
// One wave per 16x16 output tile; 4 waves/block cover the 4 N-tiles.
// Output stored channel-last f16: comp16[(b*HW + hw)*64 + oc].
// ---------------------------------------------------------------------------
__global__ void k_compress(const _Float16* __restrict__ x16,
                           const _Float16* __restrict__ wcomp16,
                           const float* __restrict__ b_comp,
                           _Float16* __restrict__ comp16) {
    const int lane = threadIdx.x & 31;
    const int wv   = threadIdx.x >> 5;          // N-tile 0..3
    const int mt   = blockIdx.x;                // 0..1023
    const int mrow = lane & 15;
    const int kb   = (lane < 16) ? 0 : 8;       // ISA 16-bit A 16x32 layout

    const int p = mt * 16 + mrow;               // A-row pixel for this lane
    const float bias = b_comp[wv * 16 + mrow];
    v8f acc;
#pragma unroll
    for (int r = 0; r < 8; ++r) acc[r] = bias;

    const _Float16* arow = x16 + (size_t)p * C_;
#pragma unroll
    for (int k0 = 0; k0 < C_; k0 += 32) {
        v8h r1 = *(const v8h*)(arow + k0 + kb);        // K = kb..kb+7
        v8h r2 = *(const v8h*)(arow + k0 + 16 + kb);   // K = 16+kb..16+kb+7
        v16h a = __builtin_shufflevector(r1, r2, 0,1,2,3,4,5,6,7,
                                                 8,9,10,11,12,13,14,15);
        v16h bm;
#pragma unroll
        for (int s = 0; s < 16; ++s)
            bm[s] = wcomp16[(size_t)(k0 + ((s < 8) ? kb + s : 16 + kb + s - 8)) * CC_ + wv * 16 + mrow];
        acc = __builtin_amdgcn_wmma_f32_16x16x32_f16(
            false, a, false, bm, (short)0, acc, false, false);
    }
    const int n   = wv * 16 + (lane & 15);
    const int mhi = (lane >= 16) ? 8 : 0;
#pragma unroll
    for (int r = 0; r < 8; ++r) {
        int pp = mt * 16 + r + mhi;
        comp16[(size_t)pp * CC_ + n] = (_Float16)acc[r];
    }
}

// ---------------------------------------------------------------------------
// Kernel 2: 5x5 encoder conv as LDS-tiled implicit GEMM.
// Block = 4x16 pixel tile (+ halo) of one image; 4 waves, one per tile row.
// A tile (8x20x64 f16 = 20KB) staged once; each 32x208 B slice (13KB)
// staged per K-step and shared by all waves/N-tiles. Each wave keeps all
// 13 N-tile accumulators so one A fragment feeds 13 WMMAs per K-step.
// ---------------------------------------------------------------------------
__global__ void __launch_bounds__(128)
k_encode(const _Float16* __restrict__ comp16,
         const _Float16* __restrict__ wenc16,
         const float* __restrict__ b_enc,
         float* __restrict__ content) {
    __shared__ __align__(128) _Float16 As[THY_ * THX_ * CC_];  // 20,480 B
    __shared__ __align__(128) _Float16 Bs[32 * ENCP_];         // 13,312 B

    const int tid  = threadIdx.x;
    const int lane = tid & 31;
    const int wv   = tid >> 5;                  // 0..3 = tile row = M-tile
    const int blk  = blockIdx.x;                // b(4) x ty(16) x tx(4)
    const int b    = blk >> 6;
    const int ty   = (blk >> 2) & 15;
    const int tx   = blk & 3;
    const int h0   = ty * TSY_, w0 = tx * TSX_;

    // --- stage A tile with halo, zero padded (v8h granularity) ---
    const v8h zz = {};
    for (int i = tid; i < THY_ * THX_ * 8; i += 128) {   // 1280 chunks
        int pi  = i >> 3;                                // 0..159
        int seg = i & 7;
        int ry = pi / THX_, rx = pi % THX_;
        int hh = h0 + ry - 2, ww = w0 + rx - 2;
        v8h v = zz;
        if (hh >= 0 && hh < H_ && ww >= 0 && ww < W_)
            v = *(const v8h*)(comp16 + ((size_t)b * HW_ + hh * W_ + ww) * CC_ + seg * 8);
        *(v8h*)(As + (size_t)pi * CC_ + seg * 8) = v;
    }

    const int mrow = lane & 15;
    const int kb   = (lane < 16) ? 0 : 8;
    const int mhi  = (lane >= 16) ? 8 : 0;

    v8f acc[13];
#pragma unroll
    for (int nt = 0; nt < 13; ++nt) {
        int n = nt * 16 + mrow;
        float bias = (n < ENC_) ? b_enc[n] : 0.f;
#pragma unroll
        for (int r = 0; r < 8; ++r) acc[nt][r] = bias;
    }

    for (int k0 = 0; k0 < KENC_; k0 += 32) {
        __syncthreads();                        // previous step's B consumed
        for (int i = tid; i < 32 * 26; i += 128) {       // 832 v8h chunks
            int row = i / 26, seg = i % 26;
            *(v8h*)(Bs + row * ENCP_ + seg * 8) =
                *(const v8h*)(wenc16 + (size_t)(k0 + row) * ENCP_ + seg * 8);
        }
        __syncthreads();

        const int tap = k0 >> 6;                // 5x5 tap index
        const int c0  = k0 & 63;                // channel sub-block 0/32
        const int dy  = tap / 5;                // 0..4 (halo-relative)
        const int dx  = tap % 5;
        const _Float16* ap =
            As + ((size_t)(wv + dy) * THX_ + (mrow + dx)) * CC_ + c0;
        v8h r1 = *(const v8h*)(ap + kb);
        v8h r2 = *(const v8h*)(ap + 16 + kb);
        v16h a = __builtin_shufflevector(r1, r2, 0,1,2,3,4,5,6,7,
                                                 8,9,10,11,12,13,14,15);
#pragma unroll
        for (int nt = 0; nt < 13; ++nt) {
            v16h bm;
#pragma unroll
            for (int s = 0; s < 16; ++s)
                bm[s] = Bs[((s < 8) ? kb + s : 16 + kb + s - 8) * ENCP_ + nt * 16 + mrow];
            acc[nt] = __builtin_amdgcn_wmma_f32_16x16x32_f16(
                false, a, false, bm, (short)0, acc[nt], false, false);
        }
    }

    // --- store: wave's M rows are pixels (h0+wv, w0+0..15) ---
#pragma unroll
    for (int nt = 0; nt < 13; ++nt) {
        int n = nt * 16 + (lane & 15);
        if (n < ENC_) {
#pragma unroll
            for (int r = 0; r < 8; ++r) {
                int m = r + mhi;
                size_t p = (size_t)b * HW_ + (size_t)(h0 + wv) * W_ + (w0 + m);
                content[p * ENC_ + n] = acc[nt][r];
            }
        }
    }
}

// ---------------------------------------------------------------------------
// Kernel 3: pixel-shuffle + softmax over the 49 channels {q*4 + sub}.
// Output kmapW[((b*HW+hw)*4 + sub)*49 + q], contiguous per output sub-pixel.
// ---------------------------------------------------------------------------
__global__ void k_softmax(const float* __restrict__ content,
                          float* __restrict__ kmapW) {
    const int idx = blockIdx.x * blockDim.x + threadIdx.x;   // 65536
    const int sub = idx & 3;
    const int bhw = idx >> 2;
    const float* row = content + (size_t)bhw * ENC_;
    float ev[K2_];
    float m = -3.4e38f;
#pragma unroll
    for (int q = 0; q < K2_; ++q) {
        float v = row[q * 4 + sub];
        ev[q] = v;
        m = fmaxf(m, v);
    }
    float s = 0.f;
#pragma unroll
    for (int q = 0; q < K2_; ++q) {
        ev[q] = __expf(ev[q] - m);
        s += ev[q];
    }
    const float inv = 1.f / s;
    float* o = kmapW + (size_t)idx * K2_;
#pragma unroll
    for (int q = 0; q < K2_; ++q) o[q] = ev[q] * inv;
}

// ---------------------------------------------------------------------------
// Kernel 4: CARAFE gather. Block = 8x8 source-pixel tile of one image.
// Kernel maps for the tile (64*196 f32 = 50KB) are fetched with gfx1250
// async global->LDS b128 loads (ASYNCcnt path, no VGPR round trip);
// x windows are staged 4 channels at a time with zero padding.
// ---------------------------------------------------------------------------
__global__ void k_carafe(const float* __restrict__ x,
                         const float* __restrict__ kmapW,
                         float* __restrict__ out) {
    __shared__ __align__(128) float km[64 * 196];   // [pix][sub*49+q]
    __shared__ __align__(128) float xs[4][14 * 14]; // 4 channels, 8+6 tile

    const int blk = blockIdx.x;         // b(4) x ty(8) x tx(8)
    const int b   = blk >> 6;
    const int ty  = (blk >> 3) & 7;
    const int tx  = blk & 7;
    const int tid = threadIdx.x;
    const int h0  = ty * 8, w0 = tx * 8;

    // Async global->LDS copy of the tile's kernel maps. Each image row of 8
    // tile pixels is 8*196 = 1568 contiguous floats in both global and LDS.
    for (int i = tid; i < 3136; i += 256) {          // 3136 float4 chunks
        int py = i / 392;                            // tile row 0..7
        int j  = i % 392;                            // float4 within row
        const float* gp = kmapW
            + ((size_t)b * HW_ + (size_t)(h0 + py) * W_ + w0) * 196 + j * 4;
        unsigned int lds_off = (unsigned int)(size_t)(km + py * 1568 + j * 4);
        asm volatile("global_load_async_to_lds_b128 %0, %1, off"
                     :: "v"(lds_off), "v"(gp) : "memory");
    }
    asm volatile("s_wait_asynccnt 0" ::: "memory");
    __syncthreads();

    const int pix = tid & 63;
    const int cl  = tid >> 6;           // 0..3
    const int py  = pix >> 3, px = pix & 7;
    const float* kmp = &km[pix * 196];

    for (int ch = 0; ch < 64; ++ch) {
        const int cbase = ch * 4;
        __syncthreads();                // previous chunk's compute done
        for (int i = tid; i < 4 * 196; i += 256) {
            int c = cbase + i / 196, s = i % 196;
            int sy = s / 14, sx = s % 14;
            int hh = h0 + sy - 3, ww = w0 + sx - 3;
            float v = 0.f;
            if (hh >= 0 && hh < H_ && ww >= 0 && ww < W_)
                v = x[((size_t)b * C_ + c) * HW_ + hh * W_ + ww];
            xs[i / 196][s] = v;
        }
        __syncthreads();

        float o00 = 0.f, o01 = 0.f, o10 = 0.f, o11 = 0.f;
        const float* xp = &xs[cl][0];
#pragma unroll
        for (int dy = 0; dy < 7; ++dy) {
#pragma unroll
            for (int dx = 0; dx < 7; ++dx) {
                const float xv = xp[(py + dy) * 14 + (px + dx)];
                const int q = dy * 7 + dx;
                o00 += xv * kmp[0 * 49 + q];
                o01 += xv * kmp[1 * 49 + q];
                o10 += xv * kmp[2 * 49 + q];
                o11 += xv * kmp[3 * 49 + q];
            }
        }
        const int c  = cbase + cl;
        const int i0 = 2 * (h0 + py), j0 = 2 * (w0 + px);
        float* ob = out + ((size_t)b * C_ + c) * OH_ * OW_;
        ob[(size_t)i0 * OW_ + j0]           = o00;
        ob[(size_t)i0 * OW_ + j0 + 1]       = o01;
        ob[(size_t)(i0 + 1) * OW_ + j0]     = o10;
        ob[(size_t)(i0 + 1) * OW_ + j0 + 1] = o11;
    }
}

// ---------------------------------------------------------------------------
extern "C" void kernel_launch(void* const* d_in, const int* in_sizes, int n_in,
                              void* d_out, int out_size, void* d_ws, size_t ws_size,
                              hipStream_t stream) {
    const float* x      = (const float*)d_in[0];
    const float* w_comp = (const float*)d_in[1];
    const float* b_comp = (const float*)d_in[2];
    const float* w_enc  = (const float*)d_in[3];
    const float* b_enc  = (const float*)d_in[4];
    float* out = (float*)d_out;

    char* ws = (char*)d_ws;
    _Float16* x16     = (_Float16*)(ws);                       //  8,388,608 B
    _Float16* comp16  = (_Float16*)(ws + 8388608);             //  2,097,152 B
    _Float16* wcomp16 = (_Float16*)(ws + 10485760);            //     32,768 B
    _Float16* wenc16  = (_Float16*)(ws + 10518528);            //    665,600 B
    float*    content = (float*)   (ws + 11184128);            // 12,845,056 B
    float*    kmapW   = (float*)   (ws + 24029184);            // 12,845,056 B

    k_prep_wcomp<<<(C_ * CC_ + 255) / 256, 256, 0, stream>>>(w_comp, wcomp16);
    k_prep_wenc<<<(KENC_ * ENCP_ + 255) / 256, 256, 0, stream>>>(w_enc, wenc16);
    k_transpose_x<<<1024, 256, 0, stream>>>(x, x16);
    k_compress<<<1024, 128, 0, stream>>>(x16, wcomp16, b_comp, comp16);
    k_encode<<<256, 128, 0, stream>>>(comp16, wenc16, b_enc, content);
    k_softmax<<<(B_ * HW_ * 4) / 256, 256, 0, stream>>>(content, kmapW);
    k_carafe<<<B_ * 64, 256, 0, stream>>>(x, kmapW, out);
}
